// LSTMModel_81140522156648
// MI455X (gfx1250) — matrix operand order repaired and checked by reading, not compiled, not run on hardware
//
#include <hip/hip_runtime.h>

// ---------------- types ----------------
typedef __attribute__((ext_vector_type(16))) __bf16          v16bf;
typedef __attribute__((ext_vector_type(16))) unsigned short  v16u;
typedef __attribute__((ext_vector_type(8)))  unsigned short  v8u;
typedef __attribute__((ext_vector_type(8)))  float           v8f;

#define HID   128
#define G4    512          // 4*HID
#define BATCH 256
#define TT    512
#define KPAD  128          // layer-0 input padded 100 -> 128

// ---------------- helpers ----------------
__device__ __forceinline__ unsigned short f2bf(float f) {
    unsigned int u = __float_as_uint(f);
    return (unsigned short)((u + 0x7FFFu + ((u >> 16) & 1u)) >> 16);   // RNE
}
__device__ __forceinline__ float bf2f(unsigned short h) {
    return __uint_as_float(((unsigned int)h) << 16);
}
// v_exp_f32 + v_rcp_f32 (TRANS pipe, co-executes with WMMA)
__device__ __forceinline__ float fast_sig(float x) {
    return __builtin_amdgcn_rcpf(1.0f + __expf(-x));
}
__device__ __forceinline__ float fast_tanh(float x) {
    return 1.0f - 2.0f * __builtin_amdgcn_rcpf(1.0f + __expf(2.0f * x));
}
// CDNA5 split workgroup barrier with LDS-only ordering: waits DScnt==0 but
// leaves global (LOADcnt/STOREcnt) traffic in flight, so the software-
// pipelined x prefetch survives the barrier.
__device__ __forceinline__ void wg_barrier_lds() {
    asm volatile("s_wait_dscnt 0x0\n\t"
                 "s_barrier_signal -1\n\t"
                 "s_barrier_wait -1" ::: "memory");
}

// ---------------- prep: x [B,T,100] f32 -> xpad [T,B,128] bf16 ----------------
__global__ void prep_x_kernel(const float* __restrict__ x, unsigned short* __restrict__ xpad) {
    int idx = blockIdx.x * blockDim.x + threadIdx.x;
    if (idx >= TT * BATCH * KPAD) return;
    int k = idx & 127;
    int b = (idx >> 7) & 255;
    int t = idx >> 15;
    float v = (k < 100) ? x[((size_t)b * TT + t) * 100 + k] : 0.0f;
    xpad[idx] = f2bf(v);
}

// ---------------- prep: weights -> bf16 (padded), bias = b_ih + b_hh ----------------
__global__ void prep_w_kernel(const float* wi0, const float* wh0, const float* bi0, const float* bh0,
                              const float* wi1, const float* wh1, const float* bi1, const float* bh1,
                              const float* wi2, const float* wh2, const float* bi2, const float* bh2,
                              const float* wi3, const float* wh3, const float* bi3, const float* bh3,
                              unsigned short* __restrict__ wih_o,   // [4][512][128] bf16
                              unsigned short* __restrict__ whh_o,   // [4][512][128] bf16
                              float* __restrict__ bias_o)           // [4][512] f32
{
    const float* WI[4] = {wi0, wi1, wi2, wi3};
    const float* WH[4] = {wh0, wh1, wh2, wh3};
    const float* BI[4] = {bi0, bi1, bi2, bi3};
    const float* BH[4] = {bh0, bh1, bh2, bh3};
    const int NW = 4 * G4 * KPAD;  // 262144
    int idx = blockIdx.x * blockDim.x + threadIdx.x;
    if (idx < NW) {
        int l = idx >> 16;
        int r = (idx >> 7) & 511;
        int k = idx & 127;
        int ind = (l == 0) ? 100 : 128;
        float vi = (k < ind) ? WI[l][r * ind + k] : 0.0f;
        wih_o[idx] = f2bf(vi);
        whh_o[idx] = f2bf(WH[l][r * 128 + k]);
    } else {
        int j = idx - NW;
        if (j < 4 * G4) {
            int l = j >> 9;
            int g = j & 511;
            bias_o[j] = BI[l][g] + BH[l][g];
        }
    }
}

// ---------------- one LSTM layer: persistent recurrence, WMMA bf16 ----------------
// grid = 16 workgroups (16 batch rows each), 256 threads (8 waves).
// Wave w owns gate N-tiles {w, w+8, w+16, w+24} = the SAME 16 columns of i,f,g,o,
// so the cell update is fully in-register; c never leaves VGPRs.
// h is double-buffered in LDS -> single barrier per timestep.
// LDS: Wih 128KB | Whh 128KB | H0 4KB | H1 4KB | bias 2KB
#define SM_WIH  0
#define SM_WHH  131072
#define SM_H0   262144
#define SM_H1   266240
#define SM_B    270336
#define SM_SIZE 272384

__global__ __launch_bounds__(256, 1)
void lstm_layer_kernel(const unsigned short* __restrict__ xin,   // [T,B,128] bf16
                       unsigned short* __restrict__ hout,        // [T,B,128] bf16
                       const unsigned short* __restrict__ wih,   // [512,128] bf16
                       const unsigned short* __restrict__ whh,   // [512,128] bf16
                       const float* __restrict__ bias)           // [512] f32
{
    extern __shared__ unsigned char smem[];
    unsigned short* sWih = (unsigned short*)(smem + SM_WIH);
    unsigned short* sWhh = (unsigned short*)(smem + SM_WHH);
    unsigned short* sH0  = (unsigned short*)(smem + SM_H0);  // [16][128] bf16
    unsigned short* sH1  = (unsigned short*)(smem + SM_H1);  // [16][128] bf16
    float*          sB   = (float*)(smem + SM_B);            // [512]

    const int tid  = threadIdx.x;
    const int wave = tid >> 5;
    const int lane = tid & 31;
    const int m16  = lane & 15;          // N for B/C/D, M for A
    const int kh   = lane >> 4;          // K-half selector
    const int b0   = blockIdx.x * 16;
    const int col  = (wave << 4) + m16;  // this lane's hidden column (0..127)

    // ---- cooperative load of weights/bias into LDS, init state ----
    {
        const uint4* gw = (const uint4*)wih;
        uint4* sw = (uint4*)sWih;
        for (int i = tid; i < (G4 * KPAD) / 8; i += 256) sw[i] = gw[i];
        gw = (const uint4*)whh; sw = (uint4*)sWhh;
        for (int i = tid; i < (G4 * KPAD) / 8; i += 256) sw[i] = gw[i];
        for (int i = tid; i < G4; i += 256) sB[i] = bias[i];
        for (int i = tid; i < 16 * HID; i += 256) { sH0[i] = 0; sH1[i] = 0; }
    }
    __syncthreads();

    // ---- loop-invariant state ----
    // W_hh B-tiles resident in registers for all 512 steps (16 x v16u = 128 VGPRs).
    v16u whr[16];
    #pragma unroll
    for (int kk = 0; kk < 4; ++kk)
        #pragma unroll
        for (int j = 0; j < 4; ++j)
            whr[kk * 4 + j] =
                *(const v16u*)(sWhh + (col + j * 128) * KPAD + kk * 32 + kh * 16);
    // Per-gate bias as ready-made WMMA C operands (seed of each acc chain).
    v8f bias_acc[4];
    #pragma unroll
    for (int j = 0; j < 4; ++j) {
        const float bvj = sB[col + j * 128];
        #pragma unroll
        for (int v = 0; v < 8; ++v) bias_acc[j][v] = bvj;
    }
    // Cell state: rows (v + 8*kh) of the batch tile, this lane's column.
    float c_reg[8];
    #pragma unroll
    for (int v = 0; v < 8; ++v) c_reg[v] = 0.0f;

    // ---- software-pipelined x loads: two register buffers, preload t=0 ----
    v8u xub[2][8];
    {
        const unsigned short* xr = xin + ((size_t)b0 + m16) * HID;
        #pragma unroll
        for (int kk = 0; kk < 4; ++kk) {
            xub[0][2 * kk]     = *(const v8u*)(xr + kk * 32 + kh * 16);
            xub[0][2 * kk + 1] = *(const v8u*)(xr + kk * 32 + 8 + kh * 16);
        }
    }

    // ---- one timestep; cur holds x_t, next t+1's x is loaded into nxt ----
    auto step = [&](int t, v8u (&cur)[8], v8u (&nxt)[8],
                    const unsigned short* sHr, unsigned short* sHw)
                    __attribute__((always_inline)) -> void {
        // A_x is a free bit-cast view of cur (cur is not overwritten this step)
        v16bf ax[4];
        #pragma unroll
        for (int kk = 0; kk < 4; ++kk) {
            v16u a;
            #pragma unroll
            for (int j = 0; j < 8; ++j) { a[j] = cur[2 * kk][j]; a[j + 8] = cur[2 * kk + 1][j]; }
            ax[kk] = __builtin_bit_cast(v16bf, a);
        }
        // issue next timestep's x loads (stay in flight across the barrier)
        if (t + 1 < TT) {
            const unsigned short* xr = xin + ((size_t)(t + 1) * BATCH + b0 + m16) * HID;
            #pragma unroll
            for (int kk = 0; kk < 4; ++kk) {
                nxt[2 * kk]     = *(const v8u*)(xr + kk * 32 + kh * 16);
                nxt[2 * kk + 1] = *(const v8u*)(xr + kk * 32 + 8 + kh * 16);
            }
        }
        // A_h = h_{t-1} from the read buffer (16x32 bf16 A, contiguous k = 16*kh + s)
        v16bf ah[4];
        {
            const unsigned short* hrow = sHr + m16 * HID;
            #pragma unroll
            for (int kk = 0; kk < 4; ++kk) {
                v8u lo = *(const v8u*)(hrow + kk * 32 + kh * 16);
                v8u hi = *(const v8u*)(hrow + kk * 32 + 8 + kh * 16);
                v16u a;
                #pragma unroll
                for (int j = 0; j < 8; ++j) { a[j] = lo[j]; a[j + 8] = hi[j]; }
                ah[kk] = __builtin_bit_cast(v16bf, a);
            }
        }

        // gates: acc[j] = bias_j + x_t@Wih_j^T + h@Whh_j^T  (j = i,f,g,o)
        v8f acc[4];
        {   // kk = 0 seeds each chain from bias via the C operand (no movs)
            v16u bw[4];
            #pragma unroll
            for (int j = 0; j < 4; ++j)
                bw[j] = *(const v16u*)(sWih + (col + j * 128) * KPAD + kh * 16);
            #pragma unroll
            for (int j = 0; j < 4; ++j)
                acc[j] = __builtin_amdgcn_wmma_f32_16x16x32_bf16(
                            false, ah[0], false, __builtin_bit_cast(v16bf, whr[j]),
                            (short)0, bias_acc[j], false, false);
            #pragma unroll
            for (int j = 0; j < 4; ++j)
                acc[j] = __builtin_amdgcn_wmma_f32_16x16x32_bf16(
                            false, ax[0], false, __builtin_bit_cast(v16bf, bw[j]),
                            (short)0, acc[j], false, false);
        }
        #pragma unroll
        for (int kk = 1; kk < 4; ++kk) {
            const int k0 = kk * 32;
            v16u bw[4];
            #pragma unroll
            for (int j = 0; j < 4; ++j)
                bw[j] = *(const v16u*)(sWih + (col + j * 128) * KPAD + k0 + kh * 16);
            #pragma unroll
            for (int j = 0; j < 4; ++j)
                acc[j] = __builtin_amdgcn_wmma_f32_16x16x32_bf16(
                            false, ah[kk], false, __builtin_bit_cast(v16bf, whr[kk * 4 + j]),
                            (short)0, acc[j], false, false);
            #pragma unroll
            for (int j = 0; j < 4; ++j)
                acc[j] = __builtin_amdgcn_wmma_f32_16x16x32_bf16(
                            false, ax[kk], false, __builtin_bit_cast(v16bf, bw[j]),
                            (short)0, acc[j], false, false);
        }

        // in-register LSTM cell update; c persists in VGPRs across all t
        {
            unsigned short* hL = sHw + col;
            unsigned short* hG = hout + ((size_t)t * BATCH + b0) * HID + col;
            const int rbase = kh * 8;
            #pragma unroll
            for (int v = 0; v < 8; ++v) {
                float i_ = acc[0][v], f_ = acc[1][v], g_ = acc[2][v], o_ = acc[3][v];
                float c  = fast_sig(f_) * c_reg[v] + fast_sig(i_) * fast_tanh(g_);
                c_reg[v] = c;
                float h  = fast_sig(o_) * fast_tanh(c);
                unsigned short hb = f2bf(h);
                const int row = rbase + v;
                hL[row * HID] = hb;
                hG[(size_t)row * HID] = hb;
            }
        }
        // single barrier per step: h_t (LDS) visible before next step's A_h reads;
        // global x prefetch stays in flight (DScnt-only wait).
        wg_barrier_lds();
    };

    for (int t = 0; t < TT; t += 2) {
        step(t,     xub[0], xub[1], sH0, sH1);
        step(t + 1, xub[1], xub[0], sH1, sH0);
    }
}

// ---------------- final FC: out[b][o] = h_last[b] . fc_w[o] + fc_b[o] ----------------
__global__ void fc_kernel(const unsigned short* __restrict__ hlast,  // [B,128] bf16
                          const float* __restrict__ fcw,             // [2,128]
                          const float* __restrict__ fcb,             // [2]
                          float* __restrict__ out)                   // [B,2]
{
    int idx = blockIdx.x * blockDim.x + threadIdx.x;
    if (idx >= BATCH * 2) return;
    int b = idx >> 1, o = idx & 1;
    float s = fcb[o];
    #pragma unroll 8
    for (int k = 0; k < HID; ++k)
        s += bf2f(hlast[b * HID + k]) * fcw[o * HID + k];
    out[idx] = s;
}

// ---------------- host launcher ----------------
extern "C" void kernel_launch(void* const* d_in, const int* in_sizes, int n_in,
                              void* d_out, int out_size, void* d_ws, size_t ws_size,
                              hipStream_t stream) {
    (void)in_sizes; (void)n_in; (void)out_size; (void)ws_size;
    const float* x = (const float*)d_in[0];
    const float* wi[4], *wh[4], *bi[4], *bh[4];
    for (int l = 0; l < 4; ++l) {
        wi[l] = (const float*)d_in[1 + 4 * l + 0];
        wh[l] = (const float*)d_in[1 + 4 * l + 1];
        bi[l] = (const float*)d_in[1 + 4 * l + 2];
        bh[l] = (const float*)d_in[1 + 4 * l + 3];
    }
    const float* fcw = (const float*)d_in[17];
    const float* fcb = (const float*)d_in[18];
    float* out = (float*)d_out;

    // workspace layout (bytes)
    unsigned char* ws = (unsigned char*)d_ws;
    const size_t SZ_ACT = (size_t)TT * BATCH * KPAD * sizeof(unsigned short);  // 32 MB
    unsigned short* xpad  = (unsigned short*)(ws);
    unsigned short* hbuf0 = (unsigned short*)(ws + SZ_ACT);
    unsigned short* hbuf1 = (unsigned short*)(ws + 2 * SZ_ACT);
    unsigned short* wihb  = (unsigned short*)(ws + 3 * SZ_ACT);                       // 4*512*128 bf16
    unsigned short* whhb  = (unsigned short*)(ws + 3 * SZ_ACT + 4 * G4 * KPAD * 2);
    float*          biasb = (float*)(ws + 3 * SZ_ACT + 2 * (4 * G4 * KPAD * 2));      // 4*512 f32

    // 1) prep x : [B,T,100] f32 -> [T,B,128] bf16
    prep_x_kernel<<<(TT * BATCH * KPAD) / 256, 256, 0, stream>>>(x, xpad);

    // 2) prep weights/biases
    {
        int total = 4 * G4 * KPAD + 4 * G4;
        prep_w_kernel<<<(total + 255) / 256, 256, 0, stream>>>(
            wi[0], wh[0], bi[0], bh[0],
            wi[1], wh[1], bi[1], bh[1],
            wi[2], wh[2], bi[2], bh[2],
            wi[3], wh[3], bi[3], bh[3],
            wihb, whhb, biasb);
    }

    // 3) 4 LSTM layers, ping-pong activation buffers
    unsigned short* lin[4]  = { xpad,  hbuf0, hbuf1, hbuf0 };
    unsigned short* lout[4] = { hbuf0, hbuf1, hbuf0, hbuf1 };
    for (int l = 0; l < 4; ++l) {
        lstm_layer_kernel<<<BATCH / 16, 256, SM_SIZE, stream>>>(
            lin[l], lout[l],
            wihb + (size_t)l * G4 * KPAD,
            whhb + (size_t)l * G4 * KPAD,
            biasb + (size_t)l * G4);
    }

    // 4) final FC from last timestep of layer 3
    const unsigned short* hlast = lout[3] + (size_t)(TT - 1) * BATCH * HID;
    fc_kernel<<<2, 256, 0, stream>>>(hlast, fcw, fcb, out);
}
